// ScaledDotProductAttention_64252710748215
// MI455X (gfx1250) — matrix operand order repaired and checked by reading, not compiled, not run on hardware
//
#include <hip/hip_runtime.h>
#include <hip/hip_bf16.h>

typedef __attribute__((ext_vector_type(16))) _Float16 v16h;
typedef __attribute__((ext_vector_type(8)))  _Float16 v8h;
typedef __attribute__((ext_vector_type(4)))  _Float16 v4h;
typedef __attribute__((ext_vector_type(8)))  float    v8f;
typedef __attribute__((ext_vector_type(4)))  float    v4f;

#define ATT_B 4
#define ATT_H 16
#define ATT_S 2048
#define ATT_D 64
#define BLOCK_Q 128      // query rows per 256-thread block
#define KEYS_PER_ITER 64
#define NWAVES 8

union V16U { v16h v; v8h h[2]; };

__device__ __forceinline__ v8f wmma_f16(v16h a, v16h b, v8f c) {
    return __builtin_amdgcn_wmma_f32_16x16x32_f16(false, a, false, b, (short)0, c, false, false);
}

// Butterfly step across the 16-lane row group (v_permlane16_b32, pure VALU).
__device__ __forceinline__ float lane16_perm(float v, unsigned lo, unsigned hi, int off) {
#if __has_builtin(__builtin_amdgcn_permlane16)
    unsigned u = __builtin_bit_cast(unsigned, v);
    u = __builtin_amdgcn_permlane16(u, u, lo, hi, false, false);
    return __builtin_bit_cast(float, u);
#else
    return __shfl_xor(v, off, 16);
#endif
}

__device__ __forceinline__ float rowmax16(float v) {
    v = fmaxf(v, lane16_perm(v, 0x67452301u, 0xEFCDAB89u, 1));
    v = fmaxf(v, lane16_perm(v, 0x54761032u, 0xDCFE98BAu, 2));
    v = fmaxf(v, lane16_perm(v, 0x32107654u, 0xBA98FEDCu, 4));
    v = fmaxf(v, lane16_perm(v, 0xFEDCBA98u, 0x76543210u, 8));
    return v;
}

__global__ __launch_bounds__(256) void
fa_fwd_kernel(const float* __restrict__ Qg, const float* __restrict__ Kg,
              const float* __restrict__ Vg, float* __restrict__ Og) {
    // Double-buffered K/V tiles + per-wave P scratch: 48 KB total.
    __shared__ _Float16 Ks[2][KEYS_PER_ITER * ATT_D];   // [buf][key][d]
    __shared__ _Float16 Vt[2][ATT_D * KEYS_PER_ITER];   // [buf][d][key]
    __shared__ _Float16 Pl[NWAVES][16 * KEYS_PER_ITER]; // per-wave [row][key]

    const int tid    = threadIdx.x;
    const int wave   = tid >> 5;
    const int lane   = tid & 31;
    const int laneHi = lane >> 4;
    const int lane15 = lane & 15;

    const int bh    = blockIdx.x;
    const int qbase = blockIdx.y * BLOCK_Q + wave * 16;

    const size_t headOff = (size_t)bh * ATT_S * ATT_D;

    // scores/d^-2 in exp2 domain; folded entirely into the Q-tile conversion.
    const float SC = 4096.0f * 1.44269504088896340736f;

    // ---- Q tile (16x64) -> A-matrix layout f16, pre-scaled by SC ----
    const float* qrow = Qg + headOff + (size_t)(qbase + lane15) * ATT_D;
    const int abase = laneHi * 16;
    v16h aQ[2];
#pragma unroll
    for (int c = 0; c < 2; ++c) {
        v4f x0 = *(const v4f*)(qrow + 32 * c + abase);
        v4f x1 = *(const v4f*)(qrow + 32 * c + abase + 4);
        v4f x2 = *(const v4f*)(qrow + 32 * c + 8 + abase);
        v4f x3 = *(const v4f*)(qrow + 32 * c + 8 + abase + 4);
        v16h a;
#pragma unroll
        for (int i = 0; i < 4; ++i) {
            a[i]      = (_Float16)(x0[i] * SC);
            a[4 + i]  = (_Float16)(x1[i] * SC);
            a[8 + i]  = (_Float16)(x2[i] * SC);
            a[12 + i] = (_Float16)(x3[i] * SC);
        }
        aQ[c] = a;
    }

    v16h onesB;
#pragma unroll
    for (int i = 0; i < 16; ++i) onesB[i] = (_Float16)1.0f;

    float mrow[8], lrow[8];
#pragma unroll
    for (int j = 0; j < 8; ++j) { mrow[j] = -1e30f; lrow[j] = 0.0f; }
    v8f acc[4];
#pragma unroll
    for (int n = 0; n < 4; ++n) acc[n] = (v8f){};

    _Float16* Pw = &Pl[wave][0];

    // Per-thread staging coordinates (float4 granularity).
    const int skey[4] = { (0 * 256 + tid) >> 4, (1 * 256 + tid) >> 4,
                          (2 * 256 + tid) >> 4, (3 * 256 + tid) >> 4 };
    const int sdp = (tid & 15) << 2;

    v4f kreg[4], vreg[4];

    // ---- software pipeline: register-stage next tile during compute ----
    auto issue_loads = [&](int kt0) {
#pragma unroll
        for (int e = 0; e < 4; ++e) {
            const float* kp = Kg + headOff + (size_t)(kt0 + skey[e]) * ATT_D + sdp;
            const float* vp = Vg + headOff + (size_t)(kt0 + skey[e]) * ATT_D + sdp;
            kreg[e] = *(const v4f*)kp;
            vreg[e] = *(const v4f*)vp;
            if (kt0 + 2 * KEYS_PER_ITER <= ATT_S - KEYS_PER_ITER) {
                __builtin_prefetch(kp + KEYS_PER_ITER * ATT_D, 0, 3);
                __builtin_prefetch(vp + KEYS_PER_ITER * ATT_D, 0, 3);
            }
        }
    };
    auto store_tile = [&](int buf) {
#pragma unroll
        for (int e = 0; e < 4; ++e) {
            v4h h;
#pragma unroll
            for (int i = 0; i < 4; ++i) h[i] = (_Float16)kreg[e][i];
            *(v4h*)&Ks[buf][skey[e] * ATT_D + sdp] = h;
#pragma unroll
            for (int i = 0; i < 4; ++i)
                Vt[buf][(sdp + i) * KEYS_PER_ITER + skey[e]] = (_Float16)vreg[e][i];
        }
    };

    const int iters = ATT_S / KEYS_PER_ITER;   // 32
    issue_loads(0);
    store_tile(0);
    __syncthreads();

    for (int it = 0; it < iters; ++it) {
        const int buf  = it & 1;
        const bool more = (it + 1 < iters);
        if (more) issue_loads((it + 1) * KEYS_PER_ITER);   // overlap with compute

        const _Float16* Kb = &Ks[buf][0];
        const _Float16* Vb = &Vt[buf][0];

        // ---- S = Q K^T for 4 key-subtiles (8 WMMAs); already exp2-domain ----
        v8f s[4];
#pragma unroll
        for (int t = 0; t < 4; ++t) {
            const int keycol = t * 16 + lane15;
            v16h bk0 = *(const v16h*)&Kb[keycol * ATT_D + laneHi * 16];
            v16h bk1 = *(const v16h*)&Kb[keycol * ATT_D + 32 + laneHi * 16];
            v8f c0 = wmma_f16(aQ[0], bk0, (v8f){});
            s[t] = wmma_f16(aQ[1], bk1, c0);
        }

        // ---- one softmax rescale per iteration ----
        float alpha[8];
#pragma unroll
        for (int j = 0; j < 8; ++j) {
            float mx = fmaxf(fmaxf(s[0][j], s[1][j]), fmaxf(s[2][j], s[3][j]));
            mx = rowmax16(mx);
            float mnew = fmaxf(mrow[j], mx);
            alpha[j] = exp2f(mrow[j] - mnew);
            mrow[j] = mnew;
        }
#pragma unroll
        for (int t = 0; t < 4; ++t)
#pragma unroll
            for (int j = 0; j < 8; ++j) {
                float p = exp2f(s[t][j] - mrow[j]);
                Pw[(j + 8 * laneHi) * KEYS_PER_ITER + t * 16 + lane15] = (_Float16)p;
            }
#pragma unroll
        for (int j = 0; j < 8; ++j) {
            lrow[j] *= alpha[j];
#pragma unroll
            for (int n = 0; n < 4; ++n) acc[n][j] *= alpha[j];
        }

        // ---- reload P in A-matrix layout (C->A relayout via LDS) ----
        v16h aP[2];
#pragma unroll
        for (int c = 0; c < 2; ++c) {
            V16U u;
            u.h[0] = *(const v8h*)&Pw[lane15 * KEYS_PER_ITER + 32 * c + abase];
            u.h[1] = *(const v8h*)&Pw[lane15 * KEYS_PER_ITER + 32 * c + 8 + abase];
            aP[c] = u.v;
        }

        // ---- row sums of P via WMMA with ones-B (2 WMMAs) ----
        v8f lsum = wmma_f16(aP[0], onesB, (v8f){});
        lsum = wmma_f16(aP[1], onesB, lsum);
#pragma unroll
        for (int j = 0; j < 8; ++j) lrow[j] += lsum[j];

        // ---- O += P V (8 WMMAs) ----
#pragma unroll
        for (int n = 0; n < 4; ++n) {
            v16h bv0 = *(const v16h*)&Vb[(16 * n + lane15) * KEYS_PER_ITER + laneHi * 16];
            v16h bv1 = *(const v16h*)&Vb[(16 * n + lane15) * KEYS_PER_ITER + 32 + laneHi * 16];
            acc[n] = wmma_f16(aP[0], bv0, acc[n]);
            acc[n] = wmma_f16(aP[1], bv1, acc[n]);
        }

        // ---- convert + stage next tile into the other buffer ----
        if (more) store_tile(buf ^ 1);
        __syncthreads();
    }

    // ---- finalize ----
    float* orow = Og + headOff + (size_t)qbase * ATT_D;
#pragma unroll
    for (int j = 0; j < 8; ++j) {
        float inv = 1.0f / lrow[j];
        int r = j + 8 * laneHi;
#pragma unroll
        for (int n = 0; n < 4; ++n)
            orow[(size_t)r * ATT_D + 16 * n + lane15] = acc[n][j] * inv;
    }
}

extern "C" void kernel_launch(void* const* d_in, const int* in_sizes, int n_in,
                              void* d_out, int out_size, void* d_ws, size_t ws_size,
                              hipStream_t stream) {
    (void)in_sizes; (void)n_in; (void)d_ws; (void)ws_size; (void)out_size;
    const float* Q = (const float*)d_in[0];
    const float* K = (const float*)d_in[1];
    const float* V = (const float*)d_in[2];
    float* O = (float*)d_out;

    dim3 grid(ATT_B * ATT_H, ATT_S / BLOCK_Q);   // (64, 16)
    dim3 block(256);
    fa_fwd_kernel<<<grid, block, 0, stream>>>(Q, K, V, O);
}
